// MultiHeadLatentAttention_28389733826627
// MI455X (gfx1250) — matrix operand order repaired and checked by reading, not compiled, not run on hardware
//
#include <hip/hip_runtime.h>
#include <hip/hip_fp16.h>
#include <math.h>

// ---------------------------------------------------------------------------
// MLA (Multi-head Latent Attention) for MI455X / gfx1250, wave32 + WMMA f16,
// pipelined Tensor Data Mover staging for the attention K tiles.
// ---------------------------------------------------------------------------

typedef _Float16 half_t;
typedef _Float16 half4 __attribute__((ext_vector_type(4)));
typedef _Float16 half8 __attribute__((ext_vector_type(8)));
typedef _Float16 v16h  __attribute__((ext_vector_type(16)));
typedef float    v8f   __attribute__((ext_vector_type(8)));
typedef unsigned int u32x4 __attribute__((ext_vector_type(4)));
typedef int          i32x4 __attribute__((ext_vector_type(4)));
typedef int          i32x8 __attribute__((ext_vector_type(8)));

#define B_   2
#define S_   2048
#define D_   2048
#define H_   16
#define DN_  128
#define DR_  64
#define DH_  128
#define QL_  768
#define KVL_ 512
#define M_   (B_ * S_)
#define DQK_ 192   // DN_ + DR_

#if defined(__HIP_DEVICE_COMPILE__) && \
    __has_builtin(__builtin_amdgcn_tensor_load_to_lds)
#define USE_TDM 1
#else
#define USE_TDM 0
#endif

__device__ __forceinline__ v16h cat16(half8 lo, half8 hi) {
  return __builtin_shufflevector(lo, hi, 0, 1, 2, 3, 4, 5, 6, 7,
                                          8, 9, 10, 11, 12, 13, 14, 15);
}

__device__ __forceinline__ v8f wmma_f16(v16h a, v16h b, v8f c) {
  return __builtin_amdgcn_wmma_f32_16x16x32_f16(
      false, a, false, b, (short)0, c, false, false);
}

#if USE_TDM
// TDM 2D tile load: global (row-major, rowLen halves per row, rowStride
// halves between rows) -> LDS, tileRows x rowLen contiguous f16.
__device__ __forceinline__ void tdm_load_2d_f16(
    const half_t* gsrc, half_t* ldst, int rowLen, int tileRows,
    int rowStride) {
  unsigned long long ga = (unsigned long long)(size_t)gsrc;
  unsigned int lo = (unsigned int)(size_t)ldst;  // addr[31:0] == LDS offset
  u32x4 g0;
  g0[0] = 1u;                                        // count=1 (valid D#)
  g0[1] = lo;                                        // lds_addr
  g0[2] = (unsigned int)(ga & 0xFFFFFFFFu);          // global_addr[31:0]
  g0[3] = (unsigned int)((ga >> 32) & 0x01FFFFFFu)   // global_addr[56:32]
          | (2u << 30);                              // type=2 ("image")
  i32x8 g1;
  g1[0] = (1 << 16);                  // workgroup_mask=0, data_size=1 (2B)
  g1[1] = (int)((unsigned)rowLen << 16);   // tensor_dim0[15:0] in [31:16]
  g1[2] = (int)((unsigned)tileRows << 16); // dim0 hi=0 | tensor_dim1 lo
  g1[3] = (int)((unsigned)rowLen << 16);   // dim1 hi=0 | tile_dim0
  g1[4] = tileRows;                   // tile_dim1 | tile_dim2=0
  g1[5] = rowStride;                  // tensor_dim0_stride[31:0]
  g1[6] = 0;                          // stride hi | tensor_dim1_stride lo
  g1[7] = 0;
  i32x4 gz = {0, 0, 0, 0};
#if __clang_major__ >= 23
  i32x8 gz8 = {0, 0, 0, 0, 0, 0, 0, 0};
  __builtin_amdgcn_tensor_load_to_lds(g0, g1, gz, gz, gz8, 0);
#else
  __builtin_amdgcn_tensor_load_to_lds(g0, g1, gz, gz, 0);
#endif
}
#endif

// ---------------------------------------------------------------------------
// f32 -> f16 conversion, float4-vectorized (all sizes are multiples of 4)
// ---------------------------------------------------------------------------
__global__ void k_f32_to_f16v(const float4* __restrict__ src,
                              half4* __restrict__ dst, long n4) {
  long i = (long)blockIdx.x * blockDim.x + threadIdx.x;
  long stride = (long)gridDim.x * blockDim.x;
  for (; i < n4; i += stride) {
    float4 v = src[i];
    half4 h;
    h[0] = (half_t)v.x; h[1] = (half_t)v.y;
    h[2] = (half_t)v.z; h[3] = (half_t)v.w;
    dst[i] = h;
  }
}

// ---------------------------------------------------------------------------
// Tiled WMMA GEMM: C[M,N] = A[M,K](f16,row) * B[K,N](f16,row)
// Block 128 thr = 4 waves (2x2), block tile 64x64, wave tile 32x32, K-step 32.
// Software-pipelined: next K-tile loaded into regs while WMMAs run from LDS.
// ---------------------------------------------------------------------------
#define GLS 40  // LDS row stride in halves (32 + 8 pad)

template <int STORE_HALF>
__global__ __launch_bounds__(128) void k_gemm_f16(
    const half_t* __restrict__ A, const half_t* __restrict__ Bm,
    float* __restrict__ Cf, half_t* __restrict__ Ch, int Ndim, int Kdim) {
  __shared__ half_t As[64 * GLS];
  __shared__ half_t Bs[64 * GLS];

  const int tid  = threadIdx.x;
  const int lane = tid & 31;
  const int w    = tid >> 5;
  const int wm   = w & 1, wn = w >> 1;
  const int lc   = lane & 15;
  const int hi   = (lane >> 4) & 1;
  const int hk   = hi * 8;
  const int kb   = hi * 16;
  const int m0   = blockIdx.y * 64;
  const int n0   = blockIdx.x * 64;

  const int idA0 = tid,          idA1 = tid + 128;
  const int rA0 = idA0 >> 2, cA0 = (idA0 & 3) << 3;
  const int rA1 = idA1 >> 2, cA1 = (idA1 & 3) << 3;
  const int rB0 = idA0 >> 3, cB0 = (idA0 & 7) << 3;
  const int rB1 = idA1 >> 3, cB1 = (idA1 & 7) << 3;

  v8f acc[2][2];
#pragma unroll
  for (int a = 0; a < 2; ++a)
#pragma unroll
    for (int c = 0; c < 2; ++c)
#pragma unroll
      for (int e = 0; e < 8; ++e) acc[a][c][e] = 0.0f;

  half8 ra0, ra1, rb0, rb1;
  ra0 = *(const half8*)(A + (size_t)(m0 + rA0) * Kdim + cA0);
  ra1 = *(const half8*)(A + (size_t)(m0 + rA1) * Kdim + cA1);
  rb0 = *(const half8*)(Bm + (size_t)rB0 * Ndim + n0 + cB0);
  rb1 = *(const half8*)(Bm + (size_t)rB1 * Ndim + n0 + cB1);

  for (int k0 = 0; k0 < Kdim; k0 += 32) {
    __syncthreads();
    *(half8*)(As + rA0 * GLS + cA0) = ra0;
    *(half8*)(As + rA1 * GLS + cA1) = ra1;
#pragma unroll
    for (int j = 0; j < 8; ++j) Bs[(cB0 + j) * GLS + rB0] = rb0[j];
#pragma unroll
    for (int j = 0; j < 8; ++j) Bs[(cB1 + j) * GLS + rB1] = rb1[j];
    __syncthreads();

    const int kn = k0 + 32;
    if (kn < Kdim) {
      ra0 = *(const half8*)(A + (size_t)(m0 + rA0) * Kdim + kn + cA0);
      ra1 = *(const half8*)(A + (size_t)(m0 + rA1) * Kdim + kn + cA1);
      rb0 = *(const half8*)(Bm + (size_t)(kn + rB0) * Ndim + n0 + cB0);
      rb1 = *(const half8*)(Bm + (size_t)(kn + rB1) * Ndim + n0 + cB1);
    }

    v16h af[2], bf[2];
#pragma unroll
    for (int a = 0; a < 2; ++a) {
      const half_t* p = As + (wm * 32 + a * 16 + lc) * GLS + kb;
      af[a] = cat16(*(const half8*)p, *(const half8*)(p + 8));
    }
#pragma unroll
    for (int c = 0; c < 2; ++c) {
      const half_t* p = Bs + (wn * 32 + c * 16 + lc) * GLS + kb;
      bf[c] = cat16(*(const half8*)p, *(const half8*)(p + 8));
    }
#pragma unroll
    for (int a = 0; a < 2; ++a)
#pragma unroll
      for (int c = 0; c < 2; ++c)
        acc[a][c] = wmma_f16(af[a], bf[c], acc[a][c]);
  }

#pragma unroll
  for (int a = 0; a < 2; ++a)
#pragma unroll
    for (int c = 0; c < 2; ++c) {
      const int row0 = m0 + wm * 32 + a * 16 + hi * 8;
      const int col  = n0 + wn * 32 + c * 16 + lc;
      if (STORE_HALF) {
        half_t* pc = Ch + (size_t)row0 * Ndim + col;
#pragma unroll
        for (int e = 0; e < 8; ++e) pc[(size_t)e * Ndim] = (half_t)acc[a][c][e];
      } else {
        float* pc = Cf + (size_t)row0 * Ndim + col;
#pragma unroll
        for (int e = 0; e < 8; ++e) pc[(size_t)e * Ndim] = acc[a][c][e];
      }
    }
}

// ---------------------------------------------------------------------------
// RMSNorm: Y[row] = X[row] * rsqrt(mean(X^2)+eps) * w, f32 in -> f16 out
// ---------------------------------------------------------------------------
__global__ __launch_bounds__(256) void k_rmsnorm_f16(
    const float* __restrict__ X, const float* __restrict__ w,
    half_t* __restrict__ Y, int C) {
  __shared__ float red[256];
  const int row = blockIdx.x;
  const float* xr = X + (size_t)row * C;
  float s = 0.f;
  for (int c = threadIdx.x; c < C; c += 256) { float v = xr[c]; s += v * v; }
  red[threadIdx.x] = s;
  __syncthreads();
  for (int off = 128; off > 0; off >>= 1) {
    if (threadIdx.x < off) red[threadIdx.x] += red[threadIdx.x + off];
    __syncthreads();
  }
  float r = rsqrtf(red[0] / (float)C + 1e-6f);
  half_t* yr = Y + (size_t)row * C;
  for (int c = threadIdx.x; c < C; c += 256)
    yr[c] = (half_t)(xr[c] * r * w[c]);
}

// ---------------------------------------------------------------------------
// RoPE + head packing: nope[M,H*128](f16) + ropePre[M,H*64](f32)
//   -> packed[B,H,S,192](f16), half-rotation convention
// ---------------------------------------------------------------------------
__global__ __launch_bounds__(256) void k_assemble_rope(
    const half_t* __restrict__ nope, const float* __restrict__ ropePre,
    const float* __restrict__ cosT, const float* __restrict__ sinT,
    half_t* __restrict__ packed) {
  const int m = blockIdx.x;
  const int b = m / S_, s = m - b * S_;
  const float* cr = cosT + (size_t)s * (DR_ / 2);
  const float* sr = sinT + (size_t)s * (DR_ / 2);
  for (int j = threadIdx.x; j < H_ * DQK_; j += 256) {
    int h = j / DQK_, d = j - h * DQK_;
    half_t val;
    if (d < DN_) {
      val = nope[(size_t)m * (H_ * DN_) + h * DN_ + d];
    } else {
      int jr = d - DN_;
      const float* rp = ropePre + (size_t)m * (H_ * DR_) + h * DR_;
      int j2 = jr & 31;
      float x1 = rp[j2], x2 = rp[32 + j2];
      float c = cr[j2], sn = sr[j2];
      float v = (jr < 32) ? (x1 * c - x2 * sn) : (x2 * c + x1 * sn);
      val = (half_t)v;
    }
    packed[(((size_t)b * H_ + h) * S_ + s) * DQK_ + d] = val;
  }
}

// ---------------------------------------------------------------------------
// Flash attention (causal, scale=1/sqrt(128)).
// Block = 128 thr = 4 waves covering 64 consecutive queries (16 per wave).
// K tiles staged by the Tensor Data Mover, double-buffered: the DMA for tile
// kt+1 is issued right after the top barrier and overlaps the entire WMMA +
// softmax body of tile kt; TENSORcnt<=1 guarantees tile kt is resident
// (tensor ops from one wave complete in order). Issue is guarded by a
// readfirstlane-uniform branch so exactly one wave runs the TDM.
//   Qp/Kp: [B,H,S,192] f16, Vh/Oh: [B,S,H*128] f16.
// ---------------------------------------------------------------------------
#define APS 40    // P tile row stride (halves)
#define VTS 40    // V^T tile row stride (halves)
#if USE_TDM
#define KTS 192   // TDM writes contiguous rows (no pad option for 96 dwords)
#define KBUF 2
#else
#define KTS 200   // 192 + 8 pad
#define KBUF 1
#endif

__global__ __launch_bounds__(128) void k_mla_attn(
    const half_t* __restrict__ Qp, const half_t* __restrict__ Kp,
    const half_t* __restrict__ Vh, half_t* __restrict__ Oh) {
  __shared__ half_t Kt[KBUF * 32 * KTS];
  __shared__ half_t Vt[128 * VTS];
  __shared__ half_t Pt[4 * 16 * APS];
  const int tid  = threadIdx.x;
  const int lane = tid & 31;
  const int w    = tid >> 5;
  const int lc   = lane & 15;
  const int hi   = (lane >> 4) & 1;
  const int hk   = hi * 8;
  const int kb   = hi * 16;
  half_t* Pl = Pt + w * 16 * APS;

  const int b = blockIdx.z;
  const int h = blockIdx.y;
  const int qbase0 = blockIdx.x * 64;
  const int qbase  = qbase0 + w * 16;
  const size_t bh = ((size_t)b * H_ + h) * S_;

  const int vkey = tid & 31, vseg = (tid >> 5) * 32;  // V: 32 keys x 128 d
#if !USE_TDM
  const int krow = tid >> 2, kseg = (tid & 3) * 48;   // K: 32 x 192
#endif

  // preload Q fragments: 6 x (16x32) covering d=0..191
  const half_t* qrow = Qp + (bh + qbase + lc) * DQK_;
  v16h qf[6];
#pragma unroll
  for (int ks = 0; ks < 6; ++ks) {
    const half_t* p = qrow + ks * 32 + kb;
    qf[ks] = cat16(*(const half8*)p, *(const half8*)(p + 8));
  }

  v8f acc[8];
  float mrow[8], lrow[8];
#pragma unroll
  for (int j = 0; j < 8; ++j)
#pragma unroll
    for (int e = 0; e < 8; ++e) acc[j][e] = 0.0f;
#pragma unroll
  for (int e = 0; e < 8; ++e) { mrow[e] = -INFINITY; lrow[e] = 0.0f; }

  const float scale = 0.08838834764831845f;  // 1/sqrt(128)
  const int ktmax = (qbase0 + 63) >> 5;      // block-uniform

#if USE_TDM
  const int wu = __builtin_amdgcn_readfirstlane(w);  // provably uniform
  if (wu == 0)  // prologue: DMA tile 0 into buffer 0
    tdm_load_2d_f16(Kp + bh * DQK_, Kt, DQK_, 32, DQK_);
#endif

  for (int kt = 0; kt <= ktmax; ++kt) {
    const int key0 = kt * 32;
#if USE_TDM
    half_t* Kcur = Kt + (kt & 1) * (32 * KTS);
#else
    half_t* Kcur = Kt;
#endif
    __syncthreads();  // previous iteration's LDS reads complete

#if USE_TDM
    if (wu == 0 && kt < ktmax)  // DMA next tile; overlaps this iteration
      tdm_load_2d_f16(Kp + (bh + key0 + 32) * DQK_,
                      Kt + ((kt + 1) & 1) * (32 * KTS), DQK_, 32, DQK_);
#else
    {
      const half_t* ksrc = Kp + (bh + key0 + krow) * DQK_ + kseg;
      half_t* kdst = Kcur + krow * KTS + kseg;
#pragma unroll
      for (int i = 0; i < 6; ++i)
        *(half8*)(kdst + i * 8) = *(const half8*)(ksrc + i * 8);
      if (kt < ktmax)
        __builtin_prefetch(Kp + (bh + key0 + 32 + krow) * DQK_ + kseg);
    }
#endif
    // ---- cooperative stage: V tile transposed -> Vt[d][key] ----
    {
      const half_t* vsrc =
          Vh + (size_t)(b * S_ + key0 + vkey) * (H_ * DH_) + h * DH_ + vseg;
#pragma unroll
      for (int i = 0; i < 4; ++i) {
        half8 vv = *(const half8*)(vsrc + i * 8);
#pragma unroll
        for (int j = 0; j < 8; ++j)
          Vt[(vseg + i * 8 + j) * VTS + vkey] = vv[j];
      }
      if (kt < ktmax)
        __builtin_prefetch(Vh + (size_t)(b * S_ + key0 + 32 + vkey) *
                                    (H_ * DH_) + h * DH_ + vseg);
    }
#if USE_TDM
    if (wu == 0) {  // current tile resident; next may stay in flight
      if (kt < ktmax) __builtin_amdgcn_s_wait_tensorcnt(1);
      else            __builtin_amdgcn_s_wait_tensorcnt(0);
    }
#endif
    __syncthreads();  // staged tiles visible

    // ---- scores S = Q K^T (two 16x16 fragments, K=192) ----
    v8f s0, s1;
#pragma unroll
    for (int e = 0; e < 8; ++e) { s0[e] = 0.0f; s1[e] = 0.0f; }
#pragma unroll
    for (int ks = 0; ks < 6; ++ks) {
      const half_t* p0 = Kcur + lc * KTS + ks * 32 + kb;
      const half_t* p1 = Kcur + (16 + lc) * KTS + ks * 32 + kb;
      v16h bf0 = cat16(*(const half8*)p0, *(const half8*)(p0 + 8));
      s0 = wmma_f16(qf[ks], bf0, s0);
      v16h bf1 = cat16(*(const half8*)p1, *(const half8*)(p1 + 8));
      s1 = wmma_f16(qf[ks], bf1, s1);
    }

    // ---- online softmax update ----
    float p0v[8], p1v[8], alpha[8];
#pragma unroll
    for (int e = 0; e < 8; ++e) {
      int rowq = qbase + e + hi * 8;
      int c0 = key0 + lc, c1 = c0 + 16;
      float v0 = (c0 <= rowq) ? s0[e] * scale : -1e30f;
      float v1 = (c1 <= rowq) ? s1[e] * scale : -1e30f;
      float mt = fmaxf(v0, v1);
      mt = fmaxf(mt, __shfl_xor(mt, 1, 32));
      mt = fmaxf(mt, __shfl_xor(mt, 2, 32));
      mt = fmaxf(mt, __shfl_xor(mt, 4, 32));
      mt = fmaxf(mt, __shfl_xor(mt, 8, 32));
      float mnew = fmaxf(mrow[e], mt);
      float a = __expf(mrow[e] - mnew);
      mrow[e] = mnew;
      alpha[e] = a;
      float e0 = __expf(v0 - mnew);
      float e1 = __expf(v1 - mnew);
      p0v[e] = e0; p1v[e] = e1;
      float rs = e0 + e1;
      rs += __shfl_xor(rs, 1, 32);
      rs += __shfl_xor(rs, 2, 32);
      rs += __shfl_xor(rs, 4, 32);
      rs += __shfl_xor(rs, 8, 32);
      lrow[e] = lrow[e] * a + rs;
    }
#pragma unroll
    for (int j = 0; j < 8; ++j)
#pragma unroll
      for (int e = 0; e < 8; ++e) acc[j][e] *= alpha[e];

    // ---- P (C-layout) -> per-wave LDS -> A-layout fragment ----
#pragma unroll
    for (int e = 0; e < 8; ++e) {
      Pl[(e + hi * 8) * APS + lc]      = (half_t)p0v[e];
      Pl[(e + hi * 8) * APS + 16 + lc] = (half_t)p1v[e];
    }
    asm volatile("" ::: "memory");  // same-wave ds ops are in-order
    const half_t* pp = Pl + lc * APS + kb;
    v16h pf = cat16(*(const half8*)pp, *(const half8*)(pp + 8));

    // ---- O += P V ----
#pragma unroll
    for (int j = 0; j < 8; ++j) {
      const half_t* vp = Vt + (j * 16 + lc) * VTS + kb;
      v16h vf = cat16(*(const half8*)vp, *(const half8*)(vp + 8));
      acc[j] = wmma_f16(pf, vf, acc[j]);
    }
    asm volatile("" ::: "memory");
  }

  // ---- epilogue: normalize and store [B,S,H*128] f16 ----
#pragma unroll
  for (int e = 0; e < 8; ++e) {
    float inv = 1.0f / lrow[e];
    int srow = qbase + e + hi * 8;
    half_t* orow = Oh + (size_t)(b * S_ + srow) * (H_ * DH_) + h * DH_;
#pragma unroll
    for (int j = 0; j < 8; ++j)
      orow[j * 16 + lc] = (half_t)(acc[j][e] * inv);
  }
}

// ---------------------------------------------------------------------------
// Host orchestration
// ---------------------------------------------------------------------------
static inline char* bump(char*& p, size_t bytes) {
  char* r = p;
  p += (bytes + 255) & ~(size_t)255;
  return r;
}

extern "C" void kernel_launch(void* const* d_in, const int* in_sizes, int n_in,
                              void* d_out, int out_size, void* d_ws,
                              size_t ws_size, hipStream_t stream) {
  (void)in_sizes; (void)n_in; (void)out_size; (void)ws_size;

  const float* x     = (const float*)d_in[0];
  const float* cosT  = (const float*)d_in[1];
  const float* sinT  = (const float*)d_in[2];
  const float* W_dq  = (const float*)d_in[3];
  const float* qnw   = (const float*)d_in[4];
  const float* W_uq  = (const float*)d_in[5];
  const float* W_qr  = (const float*)d_in[6];
  const float* W_dkv = (const float*)d_in[7];
  const float* kvnw  = (const float*)d_in[8];
  const float* W_uk  = (const float*)d_in[9];
  const float* W_kr  = (const float*)d_in[10];
  const float* W_uv  = (const float*)d_in[11];
  const float* W_o   = (const float*)d_in[12];
  float* out = (float*)d_out;

  char* p = (char*)d_ws;
  half_t* xh      = (half_t*)bump(p, (size_t)M_ * D_ * 2);
  half_t* wdq_h   = (half_t*)bump(p, (size_t)D_ * QL_ * 2);
  half_t* wuq_h   = (half_t*)bump(p, (size_t)QL_ * H_ * DN_ * 2);
  half_t* wqr_h   = (half_t*)bump(p, (size_t)QL_ * H_ * DR_ * 2);
  half_t* wdkv_h  = (half_t*)bump(p, (size_t)D_ * KVL_ * 2);
  half_t* wuk_h   = (half_t*)bump(p, (size_t)KVL_ * H_ * DN_ * 2);
  half_t* wkr_h   = (half_t*)bump(p, (size_t)KVL_ * H_ * DR_ * 2);
  half_t* wuv_h   = (half_t*)bump(p, (size_t)KVL_ * H_ * DH_ * 2);
  half_t* wo_h    = (half_t*)bump(p, (size_t)H_ * DH_ * D_ * 2);
  float*  qlat_f  = (float*)bump(p, (size_t)M_ * QL_ * 4);
  float*  kvlat_f = (float*)bump(p, (size_t)M_ * KVL_ * 4);
  half_t* qlat_h  = (half_t*)bump(p, (size_t)M_ * QL_ * 2);
  half_t* kvlat_h = (half_t*)bump(p, (size_t)M_ * KVL_ * 2);
  half_t* qnope_h = (half_t*)bump(p, (size_t)M_ * H_ * DN_ * 2);
  float*  qrope_f = (float*)bump(p, (size_t)M_ * H_ * DR_ * 4);
  half_t* knope_h = (half_t*)bump(p, (size_t)M_ * H_ * DN_ * 2);
  float*  krope_f = (float*)bump(p, (size_t)M_ * H_ * DR_ * 4);
  half_t* v_h     = (half_t*)bump(p, (size_t)M_ * H_ * DH_ * 2);
  half_t* qpack   = (half_t*)bump(p, (size_t)B_ * H_ * S_ * DQK_ * 2);
  half_t* kpack   = (half_t*)bump(p, (size_t)B_ * H_ * S_ * DQK_ * 2);
  half_t* oh      = (half_t*)bump(p, (size_t)M_ * H_ * DH_ * 2);

  const int CB = 2048;
  auto cvt = [&](const float* s, half_t* d, long n) {
    k_f32_to_f16v<<<CB, 256, 0, stream>>>((const float4*)s, (half4*)d, n / 4);
  };
  cvt(x, xh, (long)M_ * D_);
  cvt(W_dq, wdq_h, (long)D_ * QL_);
  cvt(W_uq, wuq_h, (long)QL_ * H_ * DN_);
  cvt(W_qr, wqr_h, (long)QL_ * H_ * DR_);
  cvt(W_dkv, wdkv_h, (long)D_ * KVL_);
  cvt(W_uk, wuk_h, (long)KVL_ * H_ * DN_);
  cvt(W_kr, wkr_h, (long)KVL_ * H_ * DR_);
  cvt(W_uv, wuv_h, (long)KVL_ * H_ * DH_);
  cvt(W_o, wo_h, (long)H_ * DH_ * D_);

  k_gemm_f16<0><<<dim3(QL_ / 64, M_ / 64), 128, 0, stream>>>(
      xh, wdq_h, qlat_f, nullptr, QL_, D_);
  k_gemm_f16<0><<<dim3(KVL_ / 64, M_ / 64), 128, 0, stream>>>(
      xh, wdkv_h, kvlat_f, nullptr, KVL_, D_);

  k_rmsnorm_f16<<<M_, 256, 0, stream>>>(qlat_f, qnw, qlat_h, QL_);
  k_rmsnorm_f16<<<M_, 256, 0, stream>>>(kvlat_f, kvnw, kvlat_h, KVL_);

  k_gemm_f16<1><<<dim3(H_ * DN_ / 64, M_ / 64), 128, 0, stream>>>(
      qlat_h, wuq_h, nullptr, qnope_h, H_ * DN_, QL_);
  k_gemm_f16<0><<<dim3(H_ * DR_ / 64, M_ / 64), 128, 0, stream>>>(
      qlat_h, wqr_h, qrope_f, nullptr, H_ * DR_, QL_);
  k_gemm_f16<1><<<dim3(H_ * DN_ / 64, M_ / 64), 128, 0, stream>>>(
      kvlat_h, wuk_h, nullptr, knope_h, H_ * DN_, KVL_);
  k_gemm_f16<0><<<dim3(H_ * DR_ / 64, M_ / 64), 128, 0, stream>>>(
      kvlat_h, wkr_h, krope_f, nullptr, H_ * DR_, KVL_);
  k_gemm_f16<1><<<dim3(H_ * DH_ / 64, M_ / 64), 128, 0, stream>>>(
      kvlat_h, wuv_h, nullptr, v_h, H_ * DH_, KVL_);

  k_assemble_rope<<<M_, 256, 0, stream>>>(qnope_h, qrope_f, cosT, sinT, qpack);
  k_assemble_rope<<<M_, 256, 0, stream>>>(knope_h, krope_f, cosT, sinT, kpack);

  k_mla_attn<<<dim3(S_ / 64, H_, B_), 128, 0, stream>>>(qpack, kpack, v_h, oh);

  k_gemm_f16<0><<<dim3(D_ / 64, M_ / 64), 128, 0, stream>>>(
      oh, wo_h, out, nullptr, D_, H_ * DH_);
}
